// Attention_74938589380827
// MI455X (gfx1250) — matrix operand order repaired and checked by reading, not compiled, not run on hardware
//
#include <hip/hip_runtime.h>
#include <hip/hip_bf16.h>
#include <math.h>

typedef __attribute__((ext_vector_type(8)))  float  v8f;
typedef __attribute__((ext_vector_type(8)))  __bf16 v8bf;
typedef __attribute__((ext_vector_type(16))) __bf16 v16bf;
typedef int v4i_ __attribute__((vector_size(16)));   // matches builtin's expected pointee

#define B_    2
#define N_    4096
#define DIM_  512
#define H_    8
#define DH_   64
#define W_    128
#define NW_   32
#define M_    4
#define KC_   260      // m + 2w valid key columns
#define KCP_  288      // padded to 9 chunks of 32
#define NEGF  (-3.4028234663852886e38f)

// ---- CDNA5 feature probes (device pass only; host pass falls back) ----
#if defined(__has_builtin)
#  if __has_builtin(__builtin_amdgcn_global_load_async_to_lds_b128)
#    define HAS_ASYNC 1
#  else
#    define HAS_ASYNC 0
#  endif
#  if __has_builtin(__builtin_amdgcn_tanhf)
#    define TANHF(x) __builtin_amdgcn_tanhf(x)
#  else
#    define TANHF(x) tanhf(x)
#  endif
#else
#  define HAS_ASYNC 0
#  define TANHF(x) tanhf(x)
#endif

// 16-byte global -> LDS copy; async DMA (ASYNCcnt) on CDNA5, sync fallback otherwise.
static __device__ __forceinline__ void async_cp16(const __bf16* g, __bf16* l) {
#if HAS_ASYNC
  __builtin_amdgcn_global_load_async_to_lds_b128(
      (__attribute__((address_space(1))) v4i_*)(v4i_*)g,
      (__attribute__((address_space(3))) v4i_*)(v4i_*)l, 0, 0);
#else
  *(v8bf*)l = *(const v8bf*)g;
#endif
}
static __device__ __forceinline__ void wait_async0() {
#if HAS_ASYNC
#  if __has_builtin(__builtin_amdgcn_s_wait_asynccnt)
  __builtin_amdgcn_s_wait_asynccnt(0);
#  else
  asm volatile("s_wait_asynccnt 0x0" ::: "memory");
#  endif
#endif
}

static __device__ __forceinline__ v16bf load_frag_a(const __bf16* p0, const __bf16* p1) {
  // A-matrix 16x32 bf16 lane fragment: elems 0..7 = K[koff..koff+7], 8..15 = K[koff+8..15]
  v8bf a = *(const v8bf*)p0;
  v8bf b = *(const v8bf*)p1;
  return __builtin_shufflevector(a, b, 0,1,2,3,4,5,6,7,8,9,10,11,12,13,14,15);
}

static __device__ __forceinline__ v8f wmma_bf16(v16bf a, v16bf b, v8f c) {
  return __builtin_amdgcn_wmma_f32_16x16x32_bf16(false, a, false, b, (short)0, c, false, false);
}

static __device__ __forceinline__ float rmax16(float v) {
  #pragma unroll
  for (int d = 1; d < 16; d <<= 1) v = fmaxf(v, __shfl_xor(v, d, 16));
  return v;
}
static __device__ __forceinline__ float rsum16(float v) {
  #pragma unroll
  for (int d = 1; d < 16; d <<= 1) v += __shfl_xor(v, d, 16);
  return v;
}

// ---------------------------------------------------------------------------
// Kernel 1: precision prep.  seq->bf16, Wcat = [Wq;Wkv;Wg] bf16, bcat, Wo bf16
// ---------------------------------------------------------------------------
__global__ __launch_bounds__(256) void prep_kernel(
    const float* __restrict__ seq, const float* __restrict__ Wq,
    const float* __restrict__ bq,  const float* __restrict__ Wkv,
    const float* __restrict__ Wg,  const float* __restrict__ bg,
    const float* __restrict__ Wo,
    __bf16* __restrict__ seqb, __bf16* __restrict__ Wcat,
    float* __restrict__ bcat,  __bf16* __restrict__ Wob) {
  size_t i = (size_t)blockIdx.x * 256 + threadIdx.x;
  const size_t NSEQ = (size_t)B_ * N_ * DIM_;
  const size_t NW   = (size_t)2048 * DIM_;
  const size_t NWO  = (size_t)DIM_ * DIM_;
  if (i < NSEQ) seqb[i] = (__bf16)seq[i];
  if (i < NW) {
    size_t r = i >> 9;
    float v = (r < 512) ? Wq[i] : (r < 1536) ? Wkv[i - (size_t)512 * 512]
                                             : Wg[i - (size_t)1536 * 512];
    Wcat[i] = (__bf16)v;
  }
  if (i < NWO) Wob[i] = (__bf16)Wo[i];
  if (i < 2048) bcat[i] = (i < 512) ? bq[i] : (i < 1536) ? 0.0f : bg[i - 1536];
}

// ---------------------------------------------------------------------------
// Kernel 2: fused QKVG projection GEMM. M=8192 N=2048 K=512.
// Async double-buffered LDS staging of A(128x32) and B(64x32) per K-chunk.
// ---------------------------------------------------------------------------
__global__ __launch_bounds__(256) void gemm_qkvg(
    const __bf16* __restrict__ A, const __bf16* __restrict__ Wc,
    const float* __restrict__ bias,
    __bf16* __restrict__ qo, __bf16* __restrict__ ko,
    __bf16* __restrict__ vo, float* __restrict__ go) {
  __shared__ __bf16 lA[2][128 * 32];   // 16 KB
  __shared__ __bf16 lB[2][64 * 32];    //  8 KB
  int tid = threadIdx.x;
  int wave = tid >> 5, lane = tid & 31;
  int half = lane >> 4, l16 = lane & 15;
  int koffA = half ? 16 : 0, koffB = half ? 16 : 0;
  int rowBase = blockIdx.x * 128;
  int colBase = blockIdx.y * 64;

  v8f acc[4];
  #pragma unroll
  for (int c = 0; c < 4; c++) {
    float bv = bias[colBase + c * 16 + l16];
    #pragma unroll
    for (int e = 0; e < 8; e++) acc[c][e] = bv;
  }

  auto stage = [&](int ch, int buf) {
    int kb = ch * 32;
    { // B tile: 64 cols x 32 K = 4KB, one b128 per thread
      int col = tid >> 2, part = (tid & 3) * 8;
      async_cp16(Wc + (size_t)(colBase + col) * DIM_ + kb + part,
                 &lB[buf][col * 32 + part]);
    }
    #pragma unroll
    for (int s = 0; s < 2; s++) { // A tile: 128 rows x 32 K = 8KB, two b128 per thread
      int idx = tid * 2 + s;
      int row = idx >> 2, part = (idx & 3) * 8;
      async_cp16(A + (size_t)(rowBase + row) * DIM_ + kb + part,
                 &lA[buf][row * 32 + part]);
    }
  };

  stage(0, 0);
  for (int ch = 0; ch < DIM_ / 32; ++ch) {
    int buf = ch & 1;
    wait_async0();
    __syncthreads();
    if (ch + 1 < DIM_ / 32) stage(ch + 1, buf ^ 1);
    v16bf af = load_frag_a(&lA[buf][(wave * 16 + l16) * 32 + koffA],
                           &lA[buf][(wave * 16 + l16) * 32 + koffA + 8]);
    #pragma unroll
    for (int c = 0; c < 4; c++) {
      v16bf bfv = *(const v16bf*)&lB[buf][(c * 16 + l16) * 32 + koffB];
      acc[c] = wmma_bf16(af, bfv, acc[c]);
    }
  }

  #pragma unroll
  for (int c = 0; c < 4; c++) {
    int col = colBase + c * 16 + l16;
    #pragma unroll
    for (int e = 0; e < 8; e++) {
      int r = rowBase + wave * 16 + e + half * 8;
      float v = acc[c][e];
      size_t rb = (size_t)r * DIM_;
      if      (col <  512) qo[rb + col]        = (__bf16)(v * 0.125f);  // dh^-0.5
      else if (col < 1024) ko[rb + col -  512] = (__bf16)v;
      else if (col < 1536) vo[rb + col - 1024] = (__bf16)v;
      else                 go[rb + col - 1536] = 1.0f / (1.0f + __expf(-v));
    }
  }
}

// ---------------------------------------------------------------------------
// Kernel 3: windowed attention, one workgroup per (b, h, window).
// LDS: kc[288][64] bf16 | vcT[64][288] bf16 | P[8 waves][16][32] bf16 | mc[288]
// Flash-style online softmax over 9 chunks of 32 keys; WMMA for QK^T and P*V.
// ---------------------------------------------------------------------------
__global__ __launch_bounds__(256) void attn_kernel(
    const __bf16* __restrict__ qws, const __bf16* __restrict__ kws,
    const __bf16* __restrict__ vws, const float* __restrict__ gates,
    const float* __restrict__ attn_bias, const unsigned char* __restrict__ mask,
    const unsigned char* __restrict__ wmaskp, const float* __restrict__ memkv,
    __bf16* __restrict__ gatedA) {
  extern __shared__ char smem[];
  __bf16* kcL = (__bf16*)smem;                         // [288][64]
  __bf16* vtL = (__bf16*)(smem + 36864);               // [64][288]
  __bf16* pL  = (__bf16*)(smem + 73728);               // [8][512]
  unsigned char* mcA = (unsigned char*)(smem + 81920); // [288]

  int blk = blockIdx.x;
  int x  = blk & (NW_ - 1);
  int hh = (blk >> 5) & (H_ - 1);
  int bb = blk >> 8;
  int tid = threadIdx.x;

  // ---- stage kc / vcT / key-valid mask into LDS (kc rows via async DMA) ----
  for (int j = tid; j < KCP_; j += 256) {
    if (j < M_) {
      const float* km = memkv + (((size_t)0 * H_ + hh) * M_ + j) * DH_;
      const float* vm = memkv + (((size_t)1 * H_ + hh) * M_ + j) * DH_;
      for (int d = 0; d < DH_; d++) {
        kcL[j * DH_ + d]  = (__bf16)km[d];
        vtL[d * KCP_ + j] = (__bf16)vm[d];
      }
      mcA[j] = 1;
    } else if (j < KC_) {
      int jj = j - M_;
      int wx = (jj < W_) ? (x - 1) : x;
      int off = jj & (W_ - 1);
      if (wx >= 0) {
        int p = wx * W_ + off;
        size_t base = ((size_t)(bb * N_ + p)) * DIM_ + hh * DH_;
        #pragma unroll
        for (int d4 = 0; d4 < DH_; d4 += 8)
          async_cp16(kws + base + d4, &kcL[j * DH_ + d4]);
        for (int d = 0; d < DH_; d++) vtL[d * KCP_ + j] = vws[base + d];
        mcA[j] = mask[(size_t)bb * N_ + p];
      } else {
        for (int d = 0; d < DH_; d++) { kcL[j * DH_ + d] = (__bf16)0.f; vtL[d * KCP_ + j] = (__bf16)0.f; }
        mcA[j] = 0;
      }
    } else {
      for (int d = 0; d < DH_; d++) { kcL[j * DH_ + d] = (__bf16)0.f; vtL[d * KCP_ + j] = (__bf16)0.f; }
      mcA[j] = 0;
    }
  }
  wait_async0();
  __syncthreads();

  int wave = tid >> 5, lane = tid & 31;
  int half = lane >> 4, l16 = lane & 15;
  int koffA = half ? 16 : 0, koffB = half ? 16 : 0;
  int rowi = wave * 16;

  // Q fragments for K = 0..31 and 32..63 (q pre-scaled by dh^-0.5)
  int qrow = x * W_ + rowi + l16;
  const __bf16* qp = qws + ((size_t)(bb * N_ + qrow)) * DIM_ + hh * DH_;
  v16bf qa0 = load_frag_a(qp + koffA,      qp + koffA + 8);
  v16bf qa1 = load_frag_a(qp + 32 + koffA, qp + 32 + koffA + 8);

  v8f m, l, oacc[4];
  #pragma unroll
  for (int e = 0; e < 8; e++) { m[e] = NEGF; l[e] = 0.f; }
  #pragma unroll
  for (int c = 0; c < 4; c++)
    #pragma unroll
    for (int e = 0; e < 8; e++) oacc[c][e] = 0.f;

  __bf16* pw = pL + wave * 512;

  for (int ch = 0; ch < KCP_ / 32; ch++) {
    int jbase = ch * 32;
    v8f s[2];
    #pragma unroll
    for (int t = 0; t < 2; t++) {
      #pragma unroll
      for (int e = 0; e < 8; e++) s[t][e] = 0.f;
      const __bf16* kp = kcL + (size_t)(jbase + t * 16 + l16) * DH_;
      v16bf b0 = *(const v16bf*)(kp + koffB);
      v16bf b1 = *(const v16bf*)(kp + 32 + koffB);
      s[t] = wmma_bf16(qa0, b0, s[t]);
      s[t] = wmma_bf16(qa1, b1, s[t]);
      int j = jbase + t * 16 + l16;
      bool jvalid = (mcA[j] != 0);
      #pragma unroll
      for (int e = 0; e < 8; e++) {
        int i = rowi + e + half * 8;
        float vv = s[t][e];
        if (j >= M_ && j < KC_) {
          int jj = j - M_;
          int wx = (jj < W_) ? (x - 1) : x;
          if (wx >= 0)
            vv += attn_bias[((size_t)bb * N_ + (x * W_ + i)) * N_ + (size_t)wx * W_ + (jj & (W_ - 1))];
        }
        vv = 50.f * TANHF(vv * 0.02f);
        bool ok = jvalid;
        if (ok && j >= M_)
          ok = wmaskp[(((size_t)bb * NW_ + x) * W_ + i) * (2 * W_) + (j - M_)] != 0;
        s[t][e] = ok ? vv : NEGF;
      }
    }
    // ---- online softmax update ----
    v8f mn, fac, psum;
    #pragma unroll
    for (int e = 0; e < 8; e++) {
      float rm = rmax16(fmaxf(s[0][e], s[1][e]));
      mn[e]  = fmaxf(m[e], rm);
      fac[e] = __expf(m[e] - mn[e]);
      m[e]   = mn[e];
      psum[e] = 0.f;
    }
    #pragma unroll
    for (int t = 0; t < 2; t++) {
      int col = t * 16 + l16;
      #pragma unroll
      for (int e = 0; e < 8; e++) {
        float p = __expf(s[t][e] - mn[e]);
        psum[e] += p;
        pw[(e + half * 8) * 32 + col] = (__bf16)p;   // C-layout -> LDS
      }
    }
    #pragma unroll
    for (int e = 0; e < 8; e++) {
      l[e] = l[e] * fac[e] + rsum16(psum[e]);
      #pragma unroll
      for (int c = 0; c < 4; c++) oacc[c][e] *= fac[e];
    }
    // ---- O += P * V ----
    v16bf pf = load_frag_a(pw + l16 * 32 + koffA, pw + l16 * 32 + koffA + 8);
    #pragma unroll
    for (int c = 0; c < 4; c++) {
      const __bf16* vp = vtL + (size_t)(c * 16 + l16) * KCP_ + jbase + koffB;
      v16bf vf = *(const v16bf*)vp;
      oacc[c] = wmma_bf16(pf, vf, oacc[c]);
    }
  }

  // ---- normalize, gate, store bf16 A-operand for output projection ----
  #pragma unroll
  for (int c = 0; c < 4; c++) {
    #pragma unroll
    for (int e = 0; e < 8; e++) {
      int i = rowi + e + half * 8;
      int pos = x * W_ + i;
      int dcol = c * 16 + l16;
      size_t gi = ((size_t)(bb * N_ + pos)) * DIM_ + hh * DH_ + dcol;
      float o = oacc[c][e] / l[e];
      gatedA[gi] = (__bf16)(o * gates[gi]);
    }
  }
}

// ---------------------------------------------------------------------------
// Kernel 4: output projection GEMM. M=8192 N=512 K=512, f32 out.
// Same async double-buffered LDS pipeline as kernel 2.
// ---------------------------------------------------------------------------
__global__ __launch_bounds__(256) void gemm_out(
    const __bf16* __restrict__ A, const __bf16* __restrict__ Wo,
    float* __restrict__ out) {
  __shared__ __bf16 lA[2][128 * 32];
  __shared__ __bf16 lB[2][64 * 32];
  int tid = threadIdx.x;
  int wave = tid >> 5, lane = tid & 31;
  int half = lane >> 4, l16 = lane & 15;
  int koffA = half ? 16 : 0, koffB = half ? 16 : 0;
  int rowBase = blockIdx.x * 128;
  int colBase = blockIdx.y * 64;

  v8f acc[4];
  #pragma unroll
  for (int c = 0; c < 4; c++)
    #pragma unroll
    for (int e = 0; e < 8; e++) acc[c][e] = 0.f;

  auto stage = [&](int ch, int buf) {
    int kb = ch * 32;
    { int col = tid >> 2, part = (tid & 3) * 8;
      async_cp16(Wo + (size_t)(colBase + col) * DIM_ + kb + part,
                 &lB[buf][col * 32 + part]); }
    #pragma unroll
    for (int s = 0; s < 2; s++) {
      int idx = tid * 2 + s;
      int row = idx >> 2, part = (idx & 3) * 8;
      async_cp16(A + (size_t)(rowBase + row) * DIM_ + kb + part,
                 &lA[buf][row * 32 + part]);
    }
  };

  stage(0, 0);
  for (int ch = 0; ch < DIM_ / 32; ++ch) {
    int buf = ch & 1;
    wait_async0();
    __syncthreads();
    if (ch + 1 < DIM_ / 32) stage(ch + 1, buf ^ 1);
    v16bf af = load_frag_a(&lA[buf][(wave * 16 + l16) * 32 + koffA],
                           &lA[buf][(wave * 16 + l16) * 32 + koffA + 8]);
    #pragma unroll
    for (int c = 0; c < 4; c++) {
      v16bf bfv = *(const v16bf*)&lB[buf][(c * 16 + l16) * 32 + koffB];
      acc[c] = wmma_bf16(af, bfv, acc[c]);
    }
  }

  #pragma unroll
  for (int c = 0; c < 4; c++) {
    int col = colBase + c * 16 + l16;
    #pragma unroll
    for (int e = 0; e < 8; e++) {
      int r = rowBase + wave * 16 + e + half * 8;
      out[(size_t)r * DIM_ + col] = acc[c][e];
    }
  }
}

// ---------------------------------------------------------------------------
extern "C" void kernel_launch(void* const* d_in, const int* in_sizes, int n_in,
                              void* d_out, int out_size, void* d_ws, size_t ws_size,
                              hipStream_t stream) {
  const float*         seq   = (const float*)d_in[0];
  const unsigned char* mask  = (const unsigned char*)d_in[1];
  const unsigned char* wmask = (const unsigned char*)d_in[2];
  const float*         bias  = (const float*)d_in[3];
  const float*         Wq    = (const float*)d_in[4];
  const float*         bq    = (const float*)d_in[5];
  const float*         Wkv   = (const float*)d_in[6];
  const float*         Wo    = (const float*)d_in[7];
  const float*         Wg    = (const float*)d_in[8];
  const float*         bg    = (const float*)d_in[9];
  const float*         memkv = (const float*)d_in[10];

  char* w = (char*)d_ws;
  auto take = [&](size_t bytes) { char* p = w; w += (bytes + 255) & ~(size_t)255; return p; };
  __bf16* seqb = (__bf16*)take((size_t)B_ * N_ * DIM_ * 2);
  __bf16* Wcat = (__bf16*)take((size_t)2048 * DIM_ * 2);
  float*  bcat = (float* )take((size_t)2048 * 4);
  __bf16* Wob  = (__bf16*)take((size_t)DIM_ * DIM_ * 2);
  __bf16* qws  = (__bf16*)take((size_t)B_ * N_ * DIM_ * 2);
  __bf16* kws  = (__bf16*)take((size_t)B_ * N_ * DIM_ * 2);
  __bf16* vws  = (__bf16*)take((size_t)B_ * N_ * DIM_ * 2);
  float*  gws  = (float* )take((size_t)B_ * N_ * DIM_ * 4);
  __bf16* gA   = (__bf16*)take((size_t)B_ * N_ * DIM_ * 2);

  prep_kernel<<<16384, 256, 0, stream>>>(seq, Wq, bq, Wkv, Wg, bg, Wo,
                                         seqb, Wcat, bcat, Wob);

  gemm_qkvg<<<dim3(64, 32), 256, 0, stream>>>(seqb, Wcat, bcat, qws, kws, vws, gws);

  size_t smem = 82304;  // kc 36864 + vcT 36864 + P 8192 + mc 288 (rounded)
  (void)hipFuncSetAttribute((const void*)attn_kernel,
                            hipFuncAttributeMaxDynamicSharedMemorySize, (int)smem);
  attn_kernel<<<B_ * H_ * NW_, 256, smem, stream>>>(qws, kws, vws, gws, bias,
                                                    mask, wmask, memkv, gA);

  gemm_out<<<dim3(64, 8), 256, 0, stream>>>(gA, Wob, (float*)d_out);
}
